// GraphConvolution_43748536877333
// MI455X (gfx1250) — matrix operand order repaired and checked by reading, not compiled, not run on hardware
//
#include <hip/hip_runtime.h>
#include <hip/hip_bf16.h>

// ---------- constants ----------
#define NB   32
#define CIN  64
#define COUT 128
#define TT   300
#define VV   25
#define KK_  3
#define P_TOT 240000           // NB*TT*VV
#define ROWS  9600             // NB*TT
#define KC    256              // 3*64 graph channels + 64 residual channels
#define NT   64                // positions per tile
#define XROW 1608              // padded x row stride (64*25 + 8) -> bank-conflict-free
#define XPITCH 264             // Xcat row pitch in halves (256+8) -> conflict-free b128

typedef __attribute__((ext_vector_type(16))) __bf16 bf16x16;
typedef __attribute__((ext_vector_type(8)))  float  f32x8;

union ABFrag { bf16x16 v; uint4 q[2]; };

__device__ inline __bf16 f2bf(float f) {
  unsigned u = __builtin_bit_cast(unsigned, f);
  unsigned r = u + 0x7FFFu + ((u >> 16) & 1u);           // round-to-nearest-even
  unsigned short h = (unsigned short)(r >> 16);
  return __builtin_bit_cast(__bf16, h);
}
__device__ inline float bf2f(__bf16 b) {
  unsigned u = ((unsigned)__builtin_bit_cast(unsigned short, b)) << 16;
  return __builtin_bit_cast(float, u);
}

// CDNA5 async global->LDS copy (no VGPR round-trip, tracked by ASYNCcnt)
__device__ inline void async_load_f32(const float* g, float* l) {
  asm volatile("global_load_async_to_lds_b32 %0, %1, off"
               :: "v"((unsigned)(size_t)l), "v"(g)
               : "memory");
}
__device__ inline void async_wait0() {
  asm volatile("s_wait_asynccnt 0" ::: "memory");
}

// ---------- stage 0: Ae = A*attn ; Wcat -> bf16 hi/lo ; bgsum ----------
__global__ void prep_kernel(const float* __restrict__ A, const float* __restrict__ gattn,
                            const float* __restrict__ Wg, const float* __restrict__ Wr,
                            const float* __restrict__ bg,
                            float* __restrict__ ae, __bf16* __restrict__ wh,
                            __bf16* __restrict__ wl, float* __restrict__ bgsum) {
  int gid = blockIdx.x * 256 + threadIdx.x;
  int stride = gridDim.x * 256;
  for (int e = gid; e < KK_ * VV * VV; e += stride) ae[e] = A[e] * gattn[e];
  for (int e = gid; e < COUT * KC; e += stride) {
    int o = e >> 8, kc = e & 255;
    float v = (kc < 192) ? Wg[(kc >> 6) * (COUT * CIN) + o * CIN + (kc & 63)]
                         : Wr[o * CIN + (kc - 192)];
    __bf16 h = f2bf(v);
    wh[e] = h;
    wl[e] = f2bf(v - bf2f(h));
  }
  for (int o = gid; o < COUT; o += stride)
    bgsum[o] = bg[o] + bg[COUT + o] + bg[2 * COUT + o];
}

// ---------- stage 1: fused adjacency + bf16x3 WMMA GEMM ----------
// D[o, pos] = sum_kc Wcat[o,kc] * Xcat[kc,pos];  kc<192 -> hidden, kc>=192 -> res
__global__ __launch_bounds__(512) void gemm_kernel(
    const float* __restrict__ x, const float* __restrict__ ae,
    const float* __restrict__ bgsum, const float* __restrict__ br,
    const __bf16* __restrict__ WH, const __bf16* __restrict__ WL,
    float* __restrict__ hid, float* __restrict__ res) {
  __shared__ __bf16 Xh[NT * XPITCH];    // [pos][kc] hi limb   (33792 B)
  __shared__ __bf16 Xl[NT * XPITCH];    // [pos][kc] lo limb   (33792 B)
  __shared__ float  xsF[4 * XROW];      // up to 4 (n,t) rows  (25728 B)
  __shared__ float  AeS[KK_ * VV * VV]; // (7500 B)
  __shared__ float  bgS[COUT];
  __shared__ float  brS[COUT];

  const int tid = threadIdx.x;
  const int p0  = blockIdx.x * NT;
  const int r0  = p0 / VV;

  // stage x rows for this tile via async global->LDS (a 64-pos tile spans <= 4 rows).
  // Rows beyond the true span are never read by the Xcat builder, so just clamp the
  // source address (no zero-fill needed).
  for (int e = tid; e < 4 * CIN * VV; e += 512) {
    int rr = e / (CIN * VV);
    int c  = (e / VV) % CIN;
    int v  = e % VV;
    int row = r0 + rr;
    if (row > ROWS - 1) row = ROWS - 1;
    int n = row / TT, t = row - n * TT;
    async_load_f32(x + n * (CIN * TT * VV) + c * (TT * VV) + t * VV + v,
                   xsF + rr * XROW + c * VV + v);
  }
  for (int e = tid; e < KK_ * VV * VV; e += 512) async_load_f32(ae + e, AeS + e);
  if (tid < COUT) {
    async_load_f32(bgsum + tid, bgS + tid);
    async_load_f32(br + tid, brS + tid);
  }
  async_wait0();
  __syncthreads();

  // ---- build Xcat tile: each thread owns one pos (fixed w, fixed row) x 8 channels ----
  {
    const int pos = tid & 63;
    const int cb  = (tid >> 6) * 8;
    const int pg   = p0 + pos;
    const int rowg = pg / VV;
    const int rr   = rowg - r0;
    const int w    = pg - rowg * VV;
    const float* xr0 = xsF + rr * XROW + cb * VV;

    float acc0[8], acc1[8], acc2[8];
#pragma unroll
    for (int i = 0; i < 8; ++i) { acc0[i] = 0.f; acc1[i] = 0.f; acc2[i] = 0.f; }

#pragma unroll 5
    for (int v = 0; v < VV; ++v) {
      float a0 = AeS[v * VV + w];
      float a1 = AeS[625 + v * VV + w];
      float a2 = AeS[1250 + v * VV + w];
      const float* xr = xr0 + v;
#pragma unroll
      for (int i = 0; i < 8; ++i) {
        float xv = xr[i * VV];
        acc0[i] += xv * a0;
        acc1[i] += xv * a1;
        acc2[i] += xv * a2;
      }
    }
    __bf16* xhrow = Xh + pos * XPITCH;
    __bf16* xlrow = Xl + pos * XPITCH;
#pragma unroll
    for (int i = 0; i < 8; ++i) {
      int kc = cb + i;
      __bf16 h;
      h = f2bf(acc0[i]); xhrow[kc]       = h; xlrow[kc]       = f2bf(acc0[i] - bf2f(h));
      h = f2bf(acc1[i]); xhrow[64 + kc]  = h; xlrow[64 + kc]  = f2bf(acc1[i] - bf2f(h));
      h = f2bf(acc2[i]); xhrow[128 + kc] = h; xlrow[128 + kc] = f2bf(acc2[i] - bf2f(h));
      float xi = xr0[i * VV + w];                       // residual (identity) channel
      h = f2bf(xi);      xhrow[192 + kc] = h; xlrow[192 + kc] = f2bf(xi - bf2f(h));
    }
  }
  __syncthreads();

  // ---- wave tiling: 16 waves = 4 o-rows x 4 pos-cols; each wave 32x16 out tile ----
  const int wave  = tid >> 5;
  const int lane  = tid & 31;
  const int lhalf = lane & 15;
  const int hsel  = lane >> 4;            // 0: lanes 0-15, 1: lanes 16-31
  const int obase = (wave >> 2) * 32;     // 2 M-frags per wave
  const int pfrag = (wave & 3) * 16;
  const int posL  = pfrag + lhalf;        // B col (N = lane%16)

  f32x8 acc_h0 = {0.f,0.f,0.f,0.f,0.f,0.f,0.f,0.f};
  f32x8 acc_h1 = {0.f,0.f,0.f,0.f,0.f,0.f,0.f,0.f};
  f32x8 acc_r0 = {0.f,0.f,0.f,0.f,0.f,0.f,0.f,0.f};
  f32x8 acc_r1 = {0.f,0.f,0.f,0.f,0.f,0.f,0.f,0.f};

  const __bf16* w0h = WH + (obase + lhalf) * KC;        // A row for M-frag 0
  const __bf16* w0l = WL + (obase + lhalf) * KC;
  const __bf16* w1h = WH + (obase + 16 + lhalf) * KC;   // A row for M-frag 1
  const __bf16* w1l = WL + (obase + 16 + lhalf) * KC;
  const __bf16* xhrow = Xh + posL * XPITCH;
  const __bf16* xlrow = Xl + posL * XPITCH;

#pragma unroll
  for (int kk = 0; kk < 8; ++kk) {        // 8 chunks of K=32; 0..5 hidden, 6..7 residual
    const int kcb = kk * 32;
    ABFrag a0h, a0l, a1h, a1l, bh, bl;
    // A per-lane layout: lanes<16 -> K [0..7,16..23]; lanes>=16 -> K [8..15,24..31]
    a0h.q[0] = *(const uint4*)(w0h + kcb + hsel * 8);
    a0h.q[1] = *(const uint4*)(w0h + kcb + 16 + hsel * 8);
    a0l.q[0] = *(const uint4*)(w0l + kcb + hsel * 8);
    a0l.q[1] = *(const uint4*)(w0l + kcb + 16 + hsel * 8);
    a1h.q[0] = *(const uint4*)(w1h + kcb + hsel * 8);
    a1h.q[1] = *(const uint4*)(w1h + kcb + 16 + hsel * 8);
    a1l.q[0] = *(const uint4*)(w1l + kcb + hsel * 8);
    a1l.q[1] = *(const uint4*)(w1l + kcb + 16 + hsel * 8);
    // B per-lane layout: lanes<16 -> K [0..7,16..23]; lanes>=16 -> K [8..15,24..31] (N = lane%16)
    bh.q[0] = *(const uint4*)(xhrow + kcb + hsel * 8);
    bh.q[1] = *(const uint4*)(xhrow + kcb + 16 + hsel * 8);
    bl.q[0] = *(const uint4*)(xlrow + kcb + hsel * 8);
    bl.q[1] = *(const uint4*)(xlrow + kcb + 16 + hsel * 8);

    f32x8& A0 = (kk < 6) ? acc_h0 : acc_r0;
    f32x8& A1 = (kk < 6) ? acc_h1 : acc_r1;
    A0 = __builtin_amdgcn_wmma_f32_16x16x32_bf16(false, a0h.v, false, bh.v, (short)0, A0, false, false);
    A0 = __builtin_amdgcn_wmma_f32_16x16x32_bf16(false, a0h.v, false, bl.v, (short)0, A0, false, false);
    A0 = __builtin_amdgcn_wmma_f32_16x16x32_bf16(false, a0l.v, false, bh.v, (short)0, A0, false, false);
    A1 = __builtin_amdgcn_wmma_f32_16x16x32_bf16(false, a1h.v, false, bh.v, (short)0, A1, false, false);
    A1 = __builtin_amdgcn_wmma_f32_16x16x32_bf16(false, a1h.v, false, bl.v, (short)0, A1, false, false);
    A1 = __builtin_amdgcn_wmma_f32_16x16x32_bf16(false, a1l.v, false, bh.v, (short)0, A1, false, false);
  }

  // C/D layout: VGPR i, lanes<16 -> M=i ; lanes>=16 -> M=8+i ; N = lane%16
  const int pos = p0 + posL;
#pragma unroll
  for (int i = 0; i < 8; ++i) {
    int o0 = obase + hsel * 8 + i;
    int o1 = o0 + 16;
    hid[o0 * P_TOT + pos] = acc_h0[i] + bgS[o0];
    res[o0 * P_TOT + pos] = acc_r0[i] + brS[o0];
    hid[o1 * P_TOT + pos] = acc_h1[i] + bgS[o1];
    res[o1 * P_TOT + pos] = acc_r1[i] + brS[o1];
  }
}

// ---------- stage 2: per-channel BN stats -> affine coefficients ----------
__global__ void stats_kernel(const float* __restrict__ hid, const float* __restrict__ res,
                             const float* __restrict__ g_h, const float* __restrict__ b_h,
                             const float* __restrict__ g_r, const float* __restrict__ b_r,
                             float* __restrict__ coef) {
  const int o = blockIdx.x, tid = threadIdx.x;
  const float4* h4 = (const float4*)(hid + (size_t)o * P_TOT);
  const float4* r4 = (const float4*)(res + (size_t)o * P_TOT);
  double sh = 0, ssh = 0, sr = 0, ssr = 0;
  for (int q = tid; q < P_TOT / 4; q += 256) {
    float4 h = h4[q], r = r4[q];
    sh  += (double)h.x + (double)h.y + (double)h.z + (double)h.w;
    ssh += (double)h.x * h.x + (double)h.y * h.y + (double)h.z * h.z + (double)h.w * h.w;
    sr  += (double)r.x + (double)r.y + (double)r.z + (double)r.w;
    ssr += (double)r.x * r.x + (double)r.y * r.y + (double)r.z * r.z + (double)r.w * r.w;
  }
  __shared__ double s0[256], s1[256], s2[256], s3[256];
  s0[tid] = sh; s1[tid] = ssh; s2[tid] = sr; s3[tid] = ssr;
  __syncthreads();
  for (int s = 128; s > 0; s >>= 1) {
    if (tid < s) {
      s0[tid] += s0[tid + s]; s1[tid] += s1[tid + s];
      s2[tid] += s2[tid + s]; s3[tid] += s3[tid + s];
    }
    __syncthreads();
  }
  if (tid == 0) {
    const double invP = 1.0 / (double)P_TOT;
    double mh = s0[0] * invP, vh = s1[0] * invP - mh * mh;
    double mr = s2[0] * invP, vr = s3[0] * invP - mr * mr;
    float aH = g_h[o] * rsqrtf((float)vh + 1e-5f);
    float bH = b_h[o] - (float)mh * aH;
    float aR = g_r[o] * rsqrtf((float)vr + 1e-5f);
    float bR = b_r[o] - (float)mr * aR;
    coef[o] = aH; coef[COUT + o] = bH; coef[2 * COUT + o] = aR; coef[3 * COUT + o] = bR;
  }
}

// ---------- stage 3: affine combine + layout remap to [n,o,t,v] ----------
__global__ void combine_kernel(const float* __restrict__ hid, const float* __restrict__ res,
                               const float* __restrict__ coef, float* __restrict__ out) {
  int q = blockIdx.x * 256 + threadIdx.x;          // quads; 7,680,000 total
  if (q >= (NB * COUT * (TT * VV / 4))) return;
  const int QPN = TT * VV / 4;                      // 1875 quads per (n,o)
  int no = q / QPN;
  int j  = q - no * QPN;
  int n = no >> 7, o = no & 127;
  float aH = coef[o], bH = coef[COUT + o], aR = coef[2 * COUT + o], bR = coef[3 * COUT + o];
  const float4* h4 = (const float4*)hid;
  const float4* r4 = (const float4*)res;
  float4* o4 = (float4*)out;
  int src = o * (P_TOT / 4) + n * QPN + j;          // hid/res: [o][n*7500 + ...]
  int dst = n * (COUT * QPN) + o * QPN + j;         // out: [n][o][t*V + ...]
  float4 h = h4[src], r = r4[src], v;
  v.x = h.x * aH + bH + r.x * aR + bR;
  v.y = h.y * aH + bH + r.y * aR + bR;
  v.z = h.z * aH + bH + r.z * aR + bR;
  v.w = h.w * aH + bH + r.w * aR + bR;
  o4[dst] = v;
}

// ---------- launcher ----------
extern "C" void kernel_launch(void* const* d_in, const int* in_sizes, int n_in,
                              void* d_out, int out_size, void* d_ws, size_t ws_size,
                              hipStream_t stream) {
  const float* x     = (const float*)d_in[0];
  const float* A     = (const float*)d_in[1];
  const float* gattn = (const float*)d_in[2];
  const float* Wg    = (const float*)d_in[3];
  const float* bg    = (const float*)d_in[4];
  const float* bn_g  = (const float*)d_in[5];
  const float* bn_b  = (const float*)d_in[6];
  const float* Wr    = (const float*)d_in[7];
  const float* br    = (const float*)d_in[8];
  const float* rbn_g = (const float*)d_in[9];
  const float* rbn_b = (const float*)d_in[10];
  float* out = (float*)d_out;

  // workspace layout (floats): hid | res | Ae(pad 1920) | bgsum(128) | coef(512) | Wcat hi/lo (bf16)
  float* ws    = (float*)d_ws;
  float* hid   = ws;
  float* res   = ws + (size_t)COUT * P_TOT;
  float* ae    = res + (size_t)COUT * P_TOT;
  float* bgsum = ae + 1920;
  float* coef  = bgsum + COUT;
  __bf16* wh   = (__bf16*)(coef + 4 * COUT);
  __bf16* wl   = wh + COUT * KC;

  prep_kernel<<<32, 256, 0, stream>>>(A, gattn, Wg, Wr, bg, ae, wh, wl, bgsum);
  gemm_kernel<<<P_TOT / NT, 512, 0, stream>>>(x, ae, bgsum, br, wh, wl, hid, res);
  stats_kernel<<<COUT, 256, 0, stream>>>(hid, res, bn_g, bn_b, rbn_g, rbn_b, coef);
  combine_kernel<<<(NB * COUT * (TT * VV / 4) + 255) / 256, 256, 0, stream>>>(hid, res, coef, out);
}
